// MultiHeadedAttention_36773509989178
// MI455X (gfx1250) — matrix operand order repaired and checked by reading, not compiled, not run on hardware
//
#include <hip/hip_runtime.h>

// ---------------------------------------------------------------------------
// MultiHeadedAttention for MI455X (gfx1250): bf16 WMMA everywhere,
// flash-attention online softmax with double-buffered async global->LDS
// staging of K/V tiles (ASYNCcnt path), fp32 accumulation.
//   B=2, S=2048, D=768, H=12, d_k=64
// ---------------------------------------------------------------------------

#define Bz 2
#define Sz 2048
#define Dz 768
#define Hz 12
#define DK 64
#define Mz (Bz * Sz)          // 4096 rows
#define NX (Mz * Dz)          // activation elems
#define NW (Dz * Dz)          // weight elems

typedef __bf16 bf16;
typedef __attribute__((ext_vector_type(8)))  __bf16 v8bf;
typedef __attribute__((ext_vector_type(16))) __bf16 v16bf;
typedef __attribute__((ext_vector_type(8)))  float  v8f;

// ---------------- fragment helpers (CDNA5 WMMA VGPR layouts) ----------------

__device__ __forceinline__ v16bf frag_cat(v8bf a, v8bf b) {
  v16bf r;
#pragma unroll
  for (int i = 0; i < 8; ++i) { r[i] = a[i]; r[i + 8] = b[i]; }
  return r;
}

// A-operand: 16x32 tile, row-major source with leading dim ld.
// Lane l<16 holds row l, K=0..15; lane l+16 holds K=16..31.
__device__ __forceinline__ v16bf load_fragA(const bf16* base, int ld) {
  const int lane = threadIdx.x & 31;
  const int r    = lane & 15;
  const int kh   = (lane >> 4) << 4;          // 0 or 16
  const bf16* p  = base + r * ld + kh;
  v8bf lo = *(const v8bf*)p;                  // K = kh .. kh+7
  v8bf hi = *(const v8bf*)(p + 8);            // K = kh+8 .. kh+15
  return frag_cat(lo, hi);
}

// B-operand built from an [N,K] row-major tile (computes A * B^T math).
// Lane n<16 holds col n with K=0..15; lane n+16 holds K=16..31.
__device__ __forceinline__ v16bf load_fragB_nk(const bf16* base, int ld) {
  const int lane = threadIdx.x & 31;
  const int n    = lane & 15;
  const int kb   = (lane >> 4) << 4;          // 0 or 16
  const bf16* p  = base + n * ld + kb;
  v8bf lo = *(const v8bf*)p;
  v8bf hi = *(const v8bf*)(p + 8);
  return frag_cat(lo, hi);
}

// B-operand from a [K,N] row-major tile (strided gather; used from LDS).
__device__ __forceinline__ v16bf load_fragB_kn(const bf16* base, int ld) {
  const int lane = threadIdx.x & 31;
  const int n    = lane & 15;
  const int kb   = (lane >> 4) << 4;
  v16bf f;
#pragma unroll
  for (int i = 0; i < 16; ++i) f[i] = base[(kb + i) * ld + n];
  return f;
}

__device__ __forceinline__ v8f wmma_bf16(v16bf a, v16bf b, v8f c) {
  return __builtin_amdgcn_wmma_f32_16x16x32_bf16(
      /*neg_a=*/false, a, /*neg_b=*/false, b,
      /*c_mod=*/(short)0, c, /*reuse_a=*/false, /*reuse_b=*/false);
}

// ---------------- async global->LDS (ASYNCcnt) helpers ----------------------

typedef int b128_t __attribute__((vector_size(16)));
typedef __attribute__((address_space(1))) b128_t* global_b128_ptr;
typedef __attribute__((address_space(3))) b128_t* lds_b128_ptr;

__device__ __forceinline__ void async_load_b128(const bf16* g, bf16* l) {
#if __has_builtin(__builtin_amdgcn_global_load_async_to_lds_b128)
  __builtin_amdgcn_global_load_async_to_lds_b128(
      (global_b128_ptr)(uintptr_t)g,
      (lds_b128_ptr)(unsigned)(uintptr_t)l, 0, 0);
#else
  unsigned lds_off = (unsigned)(uintptr_t)l;
  unsigned long long ga = (unsigned long long)(uintptr_t)g;
  asm volatile("global_load_async_to_lds_b128 %0, %1, off"
               :: "v"(lds_off), "v"(ga) : "memory");
#endif
}

__device__ __forceinline__ void wait_async_le4() {
#if __has_builtin(__builtin_amdgcn_s_wait_asynccnt)
  __builtin_amdgcn_s_wait_asynccnt(4);
#else
  asm volatile("s_wait_asynccnt 0x4" ::: "memory");
#endif
}

__device__ __forceinline__ void wait_async_le0() {
#if __has_builtin(__builtin_amdgcn_s_wait_asynccnt)
  __builtin_amdgcn_s_wait_asynccnt(0);
#else
  asm volatile("s_wait_asynccnt 0x0" ::: "memory");
#endif
}

// ------------------------------ kernels ------------------------------------

__global__ void cvt_f32_bf16(const float* __restrict__ in,
                             bf16* __restrict__ out, int n) {
  int i = blockIdx.x * blockDim.x + threadIdx.x;
  const int stride = gridDim.x * blockDim.x;
  for (; i < n; i += stride) out[i] = (bf16)in[i];
}

// Y = X[4096,768](bf16) * W[768,768]^T + bias, output remapped to [B,H,S,64] bf16.
// One wave computes a 16x64 output tile. 256 threads = 8 waves per block.
__global__ void proj_qkv(const bf16* __restrict__ X, const bf16* __restrict__ W,
                         const float* __restrict__ bias, bf16* __restrict__ out) {
  const int wave = blockIdx.x * (blockDim.x >> 5) + (threadIdx.x >> 5);
  const int lane = threadIdx.x & 31;
  const int mt = wave / (Dz / 64);            // 256 m-tiles
  const int nt = wave % (Dz / 64);            // 12 n64-tiles
  const int m0 = mt * 16, n0 = nt * 64;

  v8f acc[4] = {};
  for (int k = 0; k < Dz; k += 32) {
    v16bf a = load_fragA(X + (size_t)m0 * Dz + k, Dz);
    if (k + 32 < Dz) __builtin_prefetch(X + (size_t)m0 * Dz + k + 32, 0, 1);
#pragma unroll
    for (int t = 0; t < 4; ++t) {
      v16bf bf = load_fragB_nk(W + (size_t)(n0 + t * 16) * Dz + k, Dz);
      acc[t] = wmma_bf16(a, bf, acc[t]);
    }
  }
  const int n = lane & 15, half = lane >> 4;
#pragma unroll
  for (int t = 0; t < 4; ++t) {
    const int o = n0 + t * 16 + n;
    const int h = o >> 6, d = o & 63;
    const float bv = bias[o];
#pragma unroll
    for (int v = 0; v < 8; ++v) {
      const int m = m0 + half * 8 + v;
      const int bb = m >> 11, s = m & (Sz - 1);
      out[(((size_t)(bb * Hz + h)) * Sz + s) * DK + d] = (bf16)(acc[t][v] + bv);
    }
  }
}

// out[4096,768](fp32) = X(bf16) * W^T + bias
__global__ void proj_out(const bf16* __restrict__ X, const bf16* __restrict__ W,
                         const float* __restrict__ bias, float* __restrict__ out) {
  const int wave = blockIdx.x * (blockDim.x >> 5) + (threadIdx.x >> 5);
  const int lane = threadIdx.x & 31;
  const int mt = wave / (Dz / 64);
  const int nt = wave % (Dz / 64);
  const int m0 = mt * 16, n0 = nt * 64;

  v8f acc[4] = {};
  for (int k = 0; k < Dz; k += 32) {
    v16bf a = load_fragA(X + (size_t)m0 * Dz + k, Dz);
#pragma unroll
    for (int t = 0; t < 4; ++t) {
      v16bf bf = load_fragB_nk(W + (size_t)(n0 + t * 16) * Dz + k, Dz);
      acc[t] = wmma_bf16(a, bf, acc[t]);
    }
  }
  const int n = lane & 15, half = lane >> 4;
#pragma unroll
  for (int t = 0; t < 4; ++t) {
    const int o = n0 + t * 16 + n;
    const float bv = bias[o];
#pragma unroll
    for (int v = 0; v < 8; ++v) {
      const int m = m0 + half * 8 + v;
      out[(size_t)m * Dz + o] = acc[t][v] + bv;
    }
  }
}

// Flash attention: block = 4 waves = 64 query rows of one (b,h).
// K/V tiles (32x64 bf16 each) stream through LDS with double-buffered
// async global->LDS copies; all 4 waves share each staged tile.
__global__ void flash_attn(const bf16* __restrict__ Q, const bf16* __restrict__ K,
                           const bf16* __restrict__ V, const int* __restrict__ mask,
                           bf16* __restrict__ Ctx) {
  const int bh   = blockIdx.x;                // 0..23
  const int b    = bh / Hz, h = bh % Hz;
  const int tid  = threadIdx.x;               // 0..127
  const int wave = tid >> 5;                  // 0..3
  const int lane = tid & 31;
  const int n    = lane & 15, half = lane >> 4;
  const int q0   = blockIdx.y * 64 + wave * 16;

  const bf16* Qh = Q + (size_t)bh * Sz * DK;
  const bf16* Kh = K + (size_t)bh * Sz * DK;
  const bf16* Vh = V + (size_t)bh * Sz * DK;
  const float scale = 0.125f;                 // 1/sqrt(64)

  __shared__ __align__(16) bf16 Kbuf[2][32 * DK];   // 2 x 4KB
  __shared__ __align__(16) bf16 Vbuf[2][32 * DK];   // 2 x 4KB
  __shared__ __align__(16) bf16 Pbuf[4][16 * 32];   // 4KB
  bf16* myP = &Pbuf[wave][0];

  // stage one 32x64 K tile + 32x64 V tile: 128 threads x (2+2) x 16B
  auto stage = [&](int tile, int buf) {
    const bf16* gK = Kh + (size_t)tile * 32 * DK;
    const bf16* gV = Vh + (size_t)tile * 32 * DK;
#pragma unroll
    for (int r = 0; r < 2; ++r) {
      const int off = (tid + r * 128) * 8;    // 8 bf16 = 16 bytes
      async_load_b128(gK + off, &Kbuf[buf][off]);
      async_load_b128(gV + off, &Vbuf[buf][off]);
    }
  };

  const v16bf qa0 = load_fragA(Qh + (size_t)q0 * DK, DK);
  const v16bf qa1 = load_fragA(Qh + (size_t)q0 * DK + 32, DK);

  v8f oacc[4] = {};
  float mstate[8], lstate[8];
#pragma unroll
  for (int v = 0; v < 8; ++v) { mstate[v] = -3.0e38f; lstate[v] = 0.f; }

  const int NT = Sz / 32;                     // 64 key tiles
  stage(0, 0);

  for (int jt = 0; jt < NT; ++jt) {
    const int cur = jt & 1;
    if (jt + 1 < NT) {
      stage(jt + 1, (jt + 1) & 1);            // prefetch next tile
      wait_async_le4();                       // current tile's 4 loads done
    } else {
      wait_async_le0();
    }
    __syncthreads();

    const int j0 = jt * 32;
    const bf16* Kt = &Kbuf[cur][0];
    const bf16* Vt = &Vbuf[cur][0];

    // ---- scores: 16x32 = Q(16x64) * K_tile(32x64)^T (from LDS) ------------
    v8f s0 = {}, s1 = {};
    {
      v16bf kb;
      kb = load_fragB_nk(Kt, DK);                 s0 = wmma_bf16(qa0, kb, s0);
      kb = load_fragB_nk(Kt + 32, DK);            s0 = wmma_bf16(qa1, kb, s0);
      kb = load_fragB_nk(Kt + 16 * DK, DK);       s1 = wmma_bf16(qa0, kb, s1);
      kb = load_fragB_nk(Kt + 16 * DK + 32, DK);  s1 = wmma_bf16(qa1, kb, s1);
    }
    const float mk0 = (mask[b * Sz + j0 + n]      == 0) ? -1e9f : 0.f;
    const float mk1 = (mask[b * Sz + j0 + 16 + n] == 0) ? -1e9f : 0.f;

    float rmax[8];
#pragma unroll
    for (int v = 0; v < 8; ++v) {
      s0[v] = s0[v] * scale + mk0;
      s1[v] = s1[v] * scale + mk1;
      rmax[v] = fmaxf(s0[v], s1[v]);
    }
#pragma unroll
    for (int v = 0; v < 8; ++v)
#pragma unroll
      for (int xm = 1; xm < 16; xm <<= 1)
        rmax[v] = fmaxf(rmax[v], __shfl_xor(rmax[v], xm));

    // ---- online softmax ---------------------------------------------------
    float corr[8];
#pragma unroll
    for (int v = 0; v < 8; ++v) {
      const float mnew = fmaxf(mstate[v], rmax[v]);
      corr[v] = __expf(mstate[v] - mnew);
      mstate[v] = mnew;
      const float p0 = __expf(s0[v] - mnew);
      const float p1 = __expf(s1[v] - mnew);
      float rs = p0 + p1;
#pragma unroll
      for (int xm = 1; xm < 16; xm <<= 1) rs += __shfl_xor(rs, xm);
      lstate[v] = lstate[v] * corr[v] + rs;
      const int row = half * 8 + v;
      myP[row * 32 + n]      = (bf16)p0;
      myP[row * 32 + 16 + n] = (bf16)p1;
    }
#pragma unroll
    for (int t = 0; t < 4; ++t)
#pragma unroll
      for (int v = 0; v < 8; ++v) oacc[t][v] *= corr[v];

    // ---- O += P(16x32) * V_tile(32x64) (both operands in LDS) -------------
    const v16bf pa = load_fragA(myP, 32);
#pragma unroll
    for (int t = 0; t < 4; ++t) {
      v16bf vb = load_fragB_kn(Vt + t * 16, DK);
      oacc[t] = wmma_bf16(pa, vb, oacc[t]);
    }
    __syncthreads();   // all waves done with buf[cur] before it is re-staged
  }

  // ---- normalize & store context [B,S,768] --------------------------------
#pragma unroll
  for (int t = 0; t < 4; ++t) {
#pragma unroll
    for (int v = 0; v < 8; ++v) {
      const int row = q0 + half * 8 + v;
      const int col = h * DK + t * 16 + n;
      Ctx[((size_t)b * Sz + row) * Dz + col] = (bf16)(oacc[t][v] / lstate[v]);
    }
  }
}

// ------------------------------ launch -------------------------------------

extern "C" void kernel_launch(void* const* d_in, const int* in_sizes, int n_in,
                              void* d_out, int out_size, void* d_ws, size_t ws_size,
                              hipStream_t stream) {
  const float* q_in  = (const float*)d_in[0];
  const float* k_in  = (const float*)d_in[1];
  const float* v_in  = (const float*)d_in[2];
  const int*   mask  = (const int*)  d_in[3];
  const float* Wq    = (const float*)d_in[4];
  const float* bq    = (const float*)d_in[5];
  const float* Wk    = (const float*)d_in[6];
  const float* bk    = (const float*)d_in[7];
  const float* Wv    = (const float*)d_in[8];
  const float* bv    = (const float*)d_in[9];
  const float* Wo    = (const float*)d_in[10];
  const float* bo    = (const float*)d_in[11];
  float* out = (float*)d_out;

  bf16* w = (bf16*)d_ws;
  bf16* Xq  = w;                 // [4096,768]
  bf16* Xk  = w + (size_t)NX;
  bf16* Xv  = w + (size_t)2 * NX;
  bf16* Wqb = w + (size_t)3 * NX;
  bf16* Wkb = Wqb + NW;
  bf16* Wvb = Wqb + (size_t)2 * NW;
  bf16* Wob = Wqb + (size_t)3 * NW;
  bf16* Qd  = Wqb + (size_t)4 * NW;  // [B,H,S,64]
  bf16* Kd  = Qd + (size_t)NX;
  bf16* Vd  = Qd + (size_t)2 * NX;
  bf16* Ctx = Qd + (size_t)3 * NX;   // [B,S,768]

  // 1) fp32 -> bf16 conversions
  cvt_f32_bf16<<<1024, 256, 0, stream>>>(q_in, Xq, NX);
  cvt_f32_bf16<<<1024, 256, 0, stream>>>(k_in, Xk, NX);
  cvt_f32_bf16<<<1024, 256, 0, stream>>>(v_in, Xv, NX);
  cvt_f32_bf16<<<512,  256, 0, stream>>>(Wq, Wqb, NW);
  cvt_f32_bf16<<<512,  256, 0, stream>>>(Wk, Wkb, NW);
  cvt_f32_bf16<<<512,  256, 0, stream>>>(Wv, Wvb, NW);
  cvt_f32_bf16<<<512,  256, 0, stream>>>(Wo, Wob, NW);

  // 2) projections -> [B,H,S,64]; 3072 waves, 8 waves/block
  const int projBlocks = (Mz / 16) * (Dz / 64) / 8;   // 384
  proj_qkv<<<projBlocks, 256, 0, stream>>>(Xq, Wqb, bq, Qd);
  proj_qkv<<<projBlocks, 256, 0, stream>>>(Xk, Wkb, bk, Kd);
  proj_qkv<<<projBlocks, 256, 0, stream>>>(Xv, Wvb, bv, Vd);

  // 3) flash attention: 24 heads x 32 query tiles, 4 waves/block
  dim3 agrid(Bz * Hz, Sz / 64);
  flash_attn<<<agrid, 128, 0, stream>>>(Qd, Kd, Vd, mask, Ctx);

  // 4) output projection -> fp32
  proj_out<<<projBlocks, 256, 0, stream>>>(Ctx, Wob, bo, out);
}
